// DGLJTNNEncoder_5282809774597
// MI455X (gfx1250) — matrix operand order repaired and checked by reading, not compiled, not run on hardware
//
#include <hip/hip_runtime.h>
#include <hip/hip_bf16.h>
#include <stdint.h>

// ---------------- problem constants (from reference) ----------------
constexpr int Bt    = 1024;        // batch (trees)
constexpr int NT    = 32;          // nodes per tree
constexpr int Hd    = 450;         // hidden
constexpr int BN    = Bt * NT;     // 32768 nodes
constexpr int ET    = Bt * 2 * (NT - 1); // 63488 edges
constexpr int NSTEP = 10;          // 2 * max_depth
constexpr int Hp    = 480;         // padded hidden: 15 * 32 (WMMA K), mult of 16 (N)

typedef __bf16 bf16_t;
typedef bf16_t v16bf __attribute__((ext_vector_type(16)));
typedef bf16_t v8bf  __attribute__((ext_vector_type(8)));
typedef float  v8f   __attribute__((ext_vector_type(8)));

__device__ __forceinline__ float sigm(float x) { return 1.0f / (1.0f + __expf(-x)); }

// ---------------- embedding gather -> padded bf16 node matrix ----------------
__global__ __launch_bounds__(256) void k_embed(const int* wid, const float* emb,
                                               bf16_t* Xn, int n) {
    int i = blockIdx.x * blockDim.x + threadIdx.x;
    if (i >= n) return;
    int node = i / Hp, j = i % Hp;
    float v = (j < Hd) ? emb[(size_t)wid[node] * Hd + j] : 0.0f;
    Xn[i] = (bf16_t)v;
}

// ---------------- weight pack: fp32 (Hd x Hd slice) -> TRANSPOSED bf16 Hp x Hp ----
// dst[n*Hp + k] = src[(row0+k)*Hd + n]; zero-padded. N-major layout makes the
// GEMM's B-tile staging fully vectorized (b128 LDS stores, no transpose in-kernel).
__global__ __launch_bounds__(256) void k_pack_wT(const float* src, int row0, bf16_t* dst) {
    int i = blockIdx.x * blockDim.x + threadIdx.x;
    if (i >= Hp * Hp) return;
    int nn = i / Hp, k = i % Hp;
    float v = (k < Hd && nn < Hd) ? src[(size_t)(row0 + k) * Hd + nn] : 0.0f;
    dst[i] = (bf16_t)v;
}

__global__ __launch_bounds__(256) void k_zero_f32(float* p, size_t n) {
    size_t i = (size_t)blockIdx.x * blockDim.x + threadIdx.x;
    for (; i < n; i += (size_t)gridDim.x * blockDim.x) p[i] = 0.0f;
}

// ---------------- line-graph / edge segment-sum scatters ----------------
__global__ __launch_bounds__(256) void k_scatter_lg(const int* lgs, const int* lgd,
                                                    const float* m, const float* rm,
                                                    float* s, float* arm, int nlg) {
    constexpr int CH = Hp / 4; // 120 chunks of 4 floats
    long long t = (long long)blockIdx.x * blockDim.x + threadIdx.x;
    if (t >= (long long)nlg * CH) return;
    int idx = (int)(t / CH);
    int c = (int)(t % CH) * 4;
    size_t so = (size_t)lgs[idx] * Hp + c;
    size_t do_ = (size_t)lgd[idx] * Hp + c;
#pragma unroll
    for (int q = 0; q < 4; ++q) {
        atomicAdd(&s[do_ + q], m[so + q]);
        atomicAdd(&arm[do_ + q], rm[so + q]);
    }
}

__global__ __launch_bounds__(256) void k_scatter_edge(const int* edst, const float* m,
                                                      float* mn, int ne) {
    constexpr int CH = Hp / 4;
    long long t = (long long)blockIdx.x * blockDim.x + threadIdx.x;
    if (t >= (long long)ne * CH) return;
    int e = (int)(t / CH);
    int c = (int)(t % CH) * 4;
    size_t so = (size_t)e * Hp + c;
    size_t do_ = (size_t)edst[e] * Hp + c;
#pragma unroll
    for (int q = 0; q < 4; ++q) atomicAdd(&mn[do_ + q], m[so + q]);
}

// ---------------- WMMA GEMM: C[M x Hp](f32) (+)= A[M x Hp] @ W[Hp x Hp](bf16) ------
// W is stored transposed (N-major: WT[n*Hp+k]). A source: bf16 (Xn) or f32
// (activations), optional per-row gather index, optional bias, optional C accum.
// Register-staged double-buffered LDS pipeline: global fetch of tile(i+1) issues
// before the WMMAs of tile(i), LDS stores after -> loads overlap matrix math.
template <bool ABF16, bool HASIDX, bool HASBIAS, bool ACC>
__global__ __launch_bounds__(128) void k_gemm(const void* Av, const bf16_t* WT,
                                              const int* idx, const float* bias,
                                              float* C, int M) {
    constexpr int BM = 64, BNt = 48, BK = 32;
    constexpr int NIT = Hp / BK; // 15
    __shared__ bf16_t sA[2][BM][BK];   // A tile, row-major
    __shared__ bf16_t sB[2][BNt][BK];  // W tile, N-major (already transposed)

    const int tid  = threadIdx.x;
    const int lane = tid & 31;         // wave32
    const int wave = tid >> 5;         // 0..3 -> 16-row M sub-tile
    const int m0 = blockIdx.x * BM;
    const int n0 = blockIdx.y * BNt;

    // A cooperative-load mapping: thread -> (row, 16-elem half)
    const int lrow  = tid >> 1;        // 0..63
    const int lhalf = (tid & 1) * 16;  // 0 / 16
    int grow;
    if constexpr (HASIDX) grow = idx[m0 + lrow];
    else                  grow = m0 + lrow;

    // B cooperative-load mapping (96 active threads): thread -> (n row, 16-elem half)
    const int bnr   = tid >> 1;        // n row when tid < 96
    const int bhalf = (tid & 1) * 16;

    v8f acc[3];
#pragma unroll
    for (int t = 0; t < 3; ++t) acc[t] = (v8f)0.0f;

    // staging registers
    v8bf  arg[2];
    float arf[16];
    v8bf  brg[2];

    auto fetch = [&](int it) {
        const size_t ak = (size_t)grow * Hp + it * BK + lhalf;
        if constexpr (ABF16) {
            const bf16_t* src = (const bf16_t*)Av + ak;
            arg[0] = *(const v8bf*)src;
            arg[1] = *(const v8bf*)(src + 8);
        } else {
            const float* src = (const float*)Av + ak;
#pragma unroll
            for (int q = 0; q < 16; ++q) arf[q] = src[q];
        }
        if (tid < 96) {
            const bf16_t* wsrc = WT + (size_t)(n0 + bnr) * Hp + it * BK + bhalf;
            brg[0] = *(const v8bf*)wsrc;
            brg[1] = *(const v8bf*)(wsrc + 8);
        }
    };
    auto stage = [&](int buf) {
        if constexpr (ABF16) {
            *(v8bf*)&sA[buf][lrow][lhalf]     = arg[0];
            *(v8bf*)&sA[buf][lrow][lhalf + 8] = arg[1];
        } else {
            union { v8bf v[2]; bf16_t e[16]; } u;
#pragma unroll
            for (int q = 0; q < 16; ++q) u.e[q] = (bf16_t)arf[q];
            *(v8bf*)&sA[buf][lrow][lhalf]     = u.v[0];
            *(v8bf*)&sA[buf][lrow][lhalf + 8] = u.v[1];
        }
        if (tid < 96) {
            *(v8bf*)&sB[buf][bnr][bhalf]     = brg[0];
            *(v8bf*)&sB[buf][bnr][bhalf + 8] = brg[1];
        }
    };

    fetch(0);
    stage(0);
    __syncthreads();

    for (int it = 0; it < NIT; ++it) {
        const int cur = it & 1;
        const bool more = (it + 1 < NIT);
        if (more) fetch(it + 1);   // global loads in flight during WMMAs

        // A fragment: lane L -> M = L%16; K halves per 16-bit A layout
        union { v16bf v; v8bf h[2]; } fa;
        const int am = wave * 16 + (lane & 15);
        const int kb = (lane >> 4) * 8;
        fa.h[0] = *(const v8bf*)&sA[cur][am][kb];
        fa.h[1] = *(const v8bf*)&sA[cur][am][kb + 16];

#pragma unroll
        for (int nt = 0; nt < 3; ++nt) {
            // B fragment: lane L -> N = L%16, K halves per 16-bit layout (same as A)
            union { v16bf v; v8bf h[2]; } fb;
            const int bn  = nt * 16 + (lane & 15);
            const int kbb = (lane >> 4) * 8;
            fb.h[0] = *(const v8bf*)&sB[cur][bn][kbb];
            fb.h[1] = *(const v8bf*)&sB[cur][bn][kbb + 16];
            acc[nt] = __builtin_amdgcn_wmma_f32_16x16x32_bf16(
                false, fa.v, false, fb.v, (short)0, acc[nt], false, false);
        }
        if (more) stage(cur ^ 1);
        __syncthreads();
    }

    // epilogue: VGPR r -> row = r + (lane<16 ? 0 : 8); col = lane%16
#pragma unroll
    for (int nt = 0; nt < 3; ++nt) {
        const int col = n0 + nt * 16 + (lane & 15);
        float b = 0.0f;
        if constexpr (HASBIAS) b = (col < Hd) ? bias[col] : 0.0f;
#pragma unroll
        for (int r = 0; r < 8; ++r) {
            const int row = m0 + wave * 16 + r + ((lane >> 4) << 3);
            float* cp = C + (size_t)row * Hp + col;
            float v = acc[nt][r] + b;
            if constexpr (ACC) v += *cp;
            *cp = v;
        }
    }
}

// ---------------- elementwise GRU pieces ----------------
__global__ __launch_bounds__(256) void k_ew1(const float* XA, const float* ZP,
                                             const float* XH, const float* HP,
                                             const float* s, float* mnew, long long n) {
    long long i = (long long)blockIdx.x * blockDim.x + threadIdx.x;
    if (i >= n) return;
    float z = sigm(XA[i] + ZP[i]);
    float t = tanhf(XH[i] + HP[i]);
    mnew[i] = (1.0f - z) * s[i] + z * t;
}

__global__ __launch_bounds__(256) void k_ew2(const float* XR, const float* RP,
                                             const unsigned char* mask,
                                             const float* mnew, float* m, float* rm,
                                             long long n) {
    long long i = (long long)blockIdx.x * blockDim.x + threadIdx.x;
    if (i >= n) return;
    float r = sigm(XR[i] + RP[i]);
    int e = (int)(i / Hp);
    if (mask[e]) {
        float mv = mnew[i];
        m[i]  = mv;
        rm[i] = r * mv;
    }
}

__global__ __launch_bounds__(256) void k_relu_out(const float* G, float* out, int n) {
    int i = blockIdx.x * blockDim.x + threadIdx.x;
    if (i >= n) return;
    int b = i / Hd, j = i % Hd;
    out[i] = fmaxf(G[(size_t)b * Hp + j], 0.0f);
}

// ---------------- host orchestration ----------------
static inline int gcd256(long long n) { return (int)((n + 255) / 256); }

extern "C" void kernel_launch(void* const* d_in, const int* in_sizes, int n_in,
                              void* d_out, int out_size, void* d_ws, size_t ws_size,
                              hipStream_t stream) {
    const int*   wid  = (const int*)d_in[0];
    const float* emb  = (const float*)d_in[1];
    const float* Wz   = (const float*)d_in[2];
    const float* bz   = (const float*)d_in[3];
    const float* Wr   = (const float*)d_in[4];
    const float* Ur   = (const float*)d_in[5];
    const float* bU   = (const float*)d_in[6];
    const float* Wh   = (const float*)d_in[7];
    const float* bh   = (const float*)d_in[8];
    const float* Wg   = (const float*)d_in[9];
    const float* bg   = (const float*)d_in[10];
    const int*   esrc = (const int*)d_in[11];
    const int*   edst = (const int*)d_in[12];
    const int*   lgs  = (const int*)d_in[13];
    const int*   lgd  = (const int*)d_in[14];
    const unsigned char* lmask = (const unsigned char*)d_in[15];
    const int*   roots = (const int*)d_in[16];
    const int NLG = in_sizes[13];

    // workspace carve-up (base from hipMalloc is 256B-aligned; all sizes are too)
    char* p = (char*)d_ws;
    auto take = [&](size_t bytes) { char* r = p; p += bytes; return r; };
    const size_t EHp = (size_t)ET * Hp;
    bf16_t* Xn   = (bf16_t*)take((size_t)BN * Hp * 2);
    float*  XA   = (float*)take(EHp * 4);
    float*  XH   = (float*)take(EHp * 4);
    float*  XR   = (float*)take(EHp * 4);
    float*  m    = (float*)take(EHp * 4);
    float*  rm   = (float*)take(EHp * 4);
    float*  s    = (float*)take(EHp * 4);    // s & arm contiguous -> one zero call
    float*  arm  = (float*)take(EHp * 4);
    float*  mnew = (float*)take(EHp * 4);
    float*  ZP   = (float*)take(EHp * 4);    // reused as RP
    float*  HP   = (float*)take(EHp * 4);
    float*  mn   = (float*)take((size_t)BN * Hp * 4);
    float*  G    = (float*)take((size_t)Bt * Hp * 4);
    bf16_t* wzx  = (bf16_t*)take((size_t)Hp * Hp * 2);
    bf16_t* wzs  = (bf16_t*)take((size_t)Hp * Hp * 2);
    bf16_t* whx  = (bf16_t*)take((size_t)Hp * Hp * 2);
    bf16_t* whm  = (bf16_t*)take((size_t)Hp * Hp * 2);
    bf16_t* wrp  = (bf16_t*)take((size_t)Hp * Hp * 2);
    bf16_t* urp  = (bf16_t*)take((size_t)Hp * Hp * 2);
    bf16_t* wgx  = (bf16_t*)take((size_t)Hp * Hp * 2);
    bf16_t* wgm  = (bf16_t*)take((size_t)Hp * Hp * 2);
    (void)ws_size; (void)n_in; (void)out_size;

    // 1) embed + pack weights (transposed, bf16, zero-padded)
    k_embed<<<gcd256((long long)BN * Hp), 256, 0, stream>>>(wid, emb, Xn, BN * Hp);
    const int PG = gcd256((long long)Hp * Hp);
    k_pack_wT<<<PG, 256, 0, stream>>>(Wz, 0,  wzx);
    k_pack_wT<<<PG, 256, 0, stream>>>(Wz, Hd, wzs);
    k_pack_wT<<<PG, 256, 0, stream>>>(Wh, 0,  whx);
    k_pack_wT<<<PG, 256, 0, stream>>>(Wh, Hd, whm);
    k_pack_wT<<<PG, 256, 0, stream>>>(Wr, 0,  wrp);
    k_pack_wT<<<PG, 256, 0, stream>>>(Ur, 0,  urp);
    k_pack_wT<<<PG, 256, 0, stream>>>(Wg, 0,  wgx);
    k_pack_wT<<<PG, 256, 0, stream>>>(Wg, Hd, wgm);

    // 2) step-invariant precomputes: XA = src_x@Wz_top + bz, XH = src_x@Wh_top + bh,
    //    XR = dst_x@Wr + bU
    dim3 gE(ET / 64, Hp / 48);
    k_gemm<true,  true, true,  false><<<gE, 128, 0, stream>>>(Xn, wzx, esrc, bz, XA, ET);
    k_gemm<true,  true, true,  false><<<gE, 128, 0, stream>>>(Xn, whx, esrc, bh, XH, ET);
    k_gemm<true,  true, true,  false><<<gE, 128, 0, stream>>>(Xn, wrp, edst, bU, XR, ET);

    // 3) init message state
    k_zero_f32<<<4096, 256, 0, stream>>>(m,  EHp);
    k_zero_f32<<<4096, 256, 0, stream>>>(rm, EHp);

    const long long nEH = (long long)EHp;
    const int ewG = gcd256(nEH);
    const int lgG = gcd256((long long)NLG * (Hp / 4));

    // 4) level-synchronous GRU message passing
    for (int step = 0; step < NSTEP; ++step) {
        k_zero_f32<<<4096, 256, 0, stream>>>(s, 2 * EHp);   // zero s and arm
        k_scatter_lg<<<lgG, 256, 0, stream>>>(lgs, lgd, m, rm, s, arm, NLG);
        k_gemm<false, false, false, false><<<gE, 128, 0, stream>>>(s,   wzs, nullptr, nullptr, ZP, ET);
        k_gemm<false, false, false, false><<<gE, 128, 0, stream>>>(arm, whm, nullptr, nullptr, HP, ET);
        k_ew1<<<ewG, 256, 0, stream>>>(XA, ZP, XH, HP, s, mnew, nEH);
        k_gemm<false, false, false, false><<<gE, 128, 0, stream>>>(mnew, urp, nullptr, nullptr, ZP, ET);
        k_ew2<<<ewG, 256, 0, stream>>>(XR, ZP, lmask + (size_t)step * ET, mnew, m, rm, nEH);
    }

    // 5) readout: mn = segsum(m, edge_dst); h = relu([x|mn]@Wg + bg) at roots only
    k_zero_f32<<<4096, 256, 0, stream>>>(mn, (size_t)BN * Hp);
    k_scatter_edge<<<gcd256((long long)ET * (Hp / 4)), 256, 0, stream>>>(edst, m, mn, ET);
    dim3 gR(Bt / 64, Hp / 48);
    k_gemm<true,  true, true,  false><<<gR, 128, 0, stream>>>(Xn, wgx, roots, bg, G, Bt);
    k_gemm<false, true, false, true ><<<gR, 128, 0, stream>>>(mn, wgm, roots, nullptr, G, Bt);
    k_relu_out<<<gcd256((long long)Bt * Hd), 256, 0, stream>>>(G, (float*)d_out, Bt * Hd);
}
